// ConvLSTMModel_87505663689288
// MI455X (gfx1250) — matrix operand order repaired and checked by reading, not compiled, not run on hardware
//
#include <hip/hip_runtime.h>
#include <hip/hip_bf16.h>
#include <math.h>

typedef __attribute__((ext_vector_type(16))) _Float16 v16h;
typedef __attribute__((ext_vector_type(8)))  _Float16 v8h;
typedef __attribute__((ext_vector_type(8)))  float    v8f;

#define HW     65536   // 256*256
#define WIDTH  256
#define KPAD   96      // 68 real channels padded to 3*32
#define NGATES 256     // 4*64
#define HID    64
#define MTILE  32      // spatial rows per block (2 M-subtiles of 16)
#define HALO   102     // 3 intervals x 34 rows of A halo staged in LDS

// ---------------------------------------------------------------------------
// Repack W_conv [256][68][3][3] f32 -> Wt [9 taps][256 n][96 k] f16 (zero-pad k)
// ---------------------------------------------------------------------------
__global__ void clstm_prep_w(const float* __restrict__ Wc, _Float16* __restrict__ Wt) {
    int idx = blockIdx.x * 256 + threadIdx.x;          // 9*256*96 = 221184
    if (idx >= 9 * NGATES * KPAD) return;
    int k   = idx % KPAD;
    int n   = (idx / KPAD) % NGATES;
    int tap = idx / (KPAD * NGATES);
    float v = 0.0f;
    if (k < 68) v = Wc[((size_t)n * 68 + k) * 9 + tap];
    Wt[idx] = (_Float16)v;
}

// ---------------------------------------------------------------------------
// Zero c state and the h-channels (4..67) of comb0
// ---------------------------------------------------------------------------
__global__ void clstm_init(float* __restrict__ c, _Float16* __restrict__ comb0) {
    int p = blockIdx.x * 256 + threadIdx.x;
    float*     crow = c     + (size_t)p * HID;
    _Float16*  hrow = comb0 + (size_t)p * KPAD + 4;
    #pragma unroll
    for (int hc = 0; hc < HID; ++hc) { crow[hc] = 0.0f; hrow[hc] = (_Float16)0.0f; }
}

// ---------------------------------------------------------------------------
// Pack x_t (channels 0..3) into comb, zero pad channels 68..95
// ---------------------------------------------------------------------------
__global__ void clstm_pack_x(const float* __restrict__ xt, _Float16* __restrict__ comb) {
    int p = blockIdx.x * 256 + threadIdx.x;
    _Float16* row = comb + (size_t)p * KPAD;
    #pragma unroll
    for (int ch = 0; ch < 4; ++ch) row[ch] = (_Float16)xt[(size_t)ch * HW + p];
    #pragma unroll
    for (int ch = 68; ch < KPAD; ++ch) row[ch] = (_Float16)0.0f;
}

__device__ __forceinline__ float sigm(float x) { return 1.0f / (1.0f + __expf(-x)); }

// ---------------------------------------------------------------------------
// Fused conv(3x3, 96->256) + LSTM pointwise for one 32-position spatial tile.
// A halo staged in LDS (async load-to-LDS; out-of-bounds rows zero-filled at
// staging time so the hot loop is pure ds_load -> v_wmma). B from global,
// each weight element read exactly once per block. 9x3x(2Mx4N) WMMAs.
// ---------------------------------------------------------------------------
__global__ __launch_bounds__(128) void clstm_step(
    const _Float16* __restrict__ comb,      // [HW][96] f16
    const _Float16* __restrict__ Wt,        // [9][256][96] f16
    const float*    __restrict__ b_conv,    // [256] f32
    float*          __restrict__ c_state,   // [HW][64] f32
    _Float16*       __restrict__ comb_nxt)  // [HW][96] f16, h -> ch 4..67
{
    __shared__ _Float16 ldsA[HALO * KPAD];     // 19,584 B halo
    __shared__ float    lgate[MTILE * NGATES]; // 32 KB gates

    const int  tid  = threadIdx.x;
    const int  wave = tid >> 5;
    const int  lane = tid & 31;
    const bool lo   = lane < 16;
    const int  m    = lane & 15;             // A row / B column within 16-tile
    const int  tile = blockIdx.x * MTILE;

    // ---- stage A halo: 3 intervals x 34 rows. Row (iv,ro) is consumed by all
    // lanes at the SAME (yy,xx) = (tile>>8 + iv-1, (tile&255) + ro-1), so
    // boundary validity is resolved here: async-copy valid rows, zero invalid.
    {
        const int CPR   = KPAD / 8;          // 12 16B-chunks per row
        const int ytile = tile >> 8;
        const int xtile = tile & 255;
        const v8h vz = {};
        for (int i = tid; i < HALO * CPR; i += 128) {
            const int r  = i / CPR;          // halo row 0..101
            const int cc = i % CPR;
            const int iv = r / 34;           // interval = dy+1
            const int ro = r % 34;
            const int yy = ytile + iv - 1;
            const int xx = xtile + ro - 1;
            if ((yy >= 0) & (yy < WIDTH) & (xx >= 0) & (xx < WIDTH)) {
                const uint64_t src = (uint64_t)(uintptr_t)
                    (comb + ((size_t)yy * WIDTH + xx) * KPAD + cc * 8);
                const uint32_t dst = (uint32_t)(uintptr_t)(&ldsA[r * KPAD + cc * 8]);
                asm volatile("global_load_async_to_lds_b128 %0, %1, off"
                             :: "v"(dst), "v"(src) : "memory");
            } else {
                *(v8h*)(&ldsA[r * KPAD + cc * 8]) = vz;   // ds_store_b128 zeros
            }
        }
        asm volatile("s_wait_asynccnt 0x0" ::: "memory");
        __syncthreads();
    }

    // accumulators, bias folded in (all elements of one tile share N = ncol)
    v8f acc[2][4];
    #pragma unroll
    for (int j = 0; j < 4; ++j) {
        const float bj = b_conv[(wave * 4 + j) * 16 + m];
        const v8f bsp = { bj, bj, bj, bj, bj, bj, bj, bj };
        acc[0][j] = bsp;
        acc[1][j] = bsp;
    }

    #pragma unroll
    for (int tap = 0; tap < 9; ++tap) {
        const int dy = tap / 3 - 1;          // compile-time constants
        const int dx = tap % 3 - 1;

        #pragma unroll
        for (int kb = 0; kb < 3; ++kb) {
            // ---- A fragments (16x32) straight from LDS, no masking needed:
            // lanes 0-15 K{0..7,16..23}, lanes 16-31 K{8..15,24..31}
            const int k0 = kb * 32 + (lo ? 0 : 8);
            v16h afrag[2];
            #pragma unroll
            for (int ms = 0; ms < 2; ++ms) {
                const int lr = (dy + 1) * 34 + ms * 16 + m + dx + 1;
                const _Float16* arow = &ldsA[lr * KPAD];
                v8h a0 = *(const v8h*)(arow + k0);       // ds_load_b128
                v8h a1 = *(const v8h*)(arow + k0 + 16);
                afrag[ms] = __builtin_shufflevector(a0, a1,
                    0,1,2,3,4,5,6,7,8,9,10,11,12,13,14,15);
            }

            // ---- all 4 B fragments first (one load clause, max distance to use)
            v16h bfrag[4];
            #pragma unroll
            for (int j = 0; j < 4; ++j) {
                const int n = (wave * 4 + j) * 16 + m;
                const _Float16* brow =
                    Wt + ((size_t)tap * NGATES + n) * KPAD + kb * 32 + (lo ? 0 : 8);
                v8h b0 = *(const v8h*)(brow);
                v8h b1 = *(const v8h*)(brow + 16);
                bfrag[j] = __builtin_shufflevector(b0, b1,
                    0,1,2,3,4,5,6,7,8,9,10,11,12,13,14,15);
            }

            // ---- 8 WMMAs for this k-step
            #pragma unroll
            for (int j = 0; j < 4; ++j)
                #pragma unroll
                for (int ms = 0; ms < 2; ++ms)
                    acc[ms][j] = __builtin_amdgcn_wmma_f32_16x16x32_f16(
                        false, afrag[ms], false, bfrag[j],
                        (short)0, acc[ms][j], false, false);
        }
    }

    // ---- spill gates (with bias) to LDS; C/D layout: VGPR r -> M = r + 8*hi ----
    const int mrow = lo ? 0 : 8;
    #pragma unroll
    for (int ms = 0; ms < 2; ++ms) {
        #pragma unroll
        for (int j = 0; j < 4; ++j) {
            const int ncol = (wave * 4 + j) * 16 + m;
            #pragma unroll
            for (int r = 0; r < 8; ++r)
                lgate[(ms * 16 + mrow + r) * NGATES + ncol] = acc[ms][j][r];
        }
    }
    __syncthreads();

    // ---- fused LSTM pointwise: 32 positions x 64 hidden channels ----
    for (int it = tid; it < MTILE * HID; it += 128) {
        const int mm = it >> 6;
        const int hc = it & 63;
        const int pp = tile + mm;
        const float* lg = lgate + mm * NGATES;
        const float gi = lg[hc];
        const float gf = lg[HID + hc];
        const float go = lg[2 * HID + hc];
        const float gg = lg[3 * HID + hc];
        const float co = c_state[(size_t)pp * HID + hc];
        const float cn = sigm(gf) * co + sigm(gi) * tanhf(gg);
        const float hn = sigm(go) * tanhf(cn);
        c_state[(size_t)pp * HID + hc] = cn;
        comb_nxt[(size_t)pp * KPAD + 4 + hc] = (_Float16)hn;
    }
}

// ---------------------------------------------------------------------------
// 1x1 output conv: out[p] = b_out + sum_hc W_out[hc] * h[p][hc]
// ---------------------------------------------------------------------------
__global__ void clstm_out(const _Float16* __restrict__ comb,
                          const float* __restrict__ W_out,
                          const float* __restrict__ b_out,
                          float* __restrict__ out) {
    int p = blockIdx.x * 256 + threadIdx.x;
    const _Float16* hrow = comb + (size_t)p * KPAD + 4;
    float s = b_out[0];
    #pragma unroll
    for (int hc = 0; hc < HID; ++hc) s += W_out[hc] * (float)hrow[hc];
    out[p] = s;
}

// ---------------------------------------------------------------------------
extern "C" void kernel_launch(void* const* d_in, const int* in_sizes, int n_in,
                              void* d_out, int out_size, void* d_ws, size_t ws_size,
                              hipStream_t stream) {
    const float* seq    = (const float*)d_in[0];   // [16][4][256][256]
    const float* W_conv = (const float*)d_in[1];   // [256][68][3][3]
    const float* b_conv = (const float*)d_in[2];   // [256]
    const float* W_out  = (const float*)d_in[3];   // [64]
    const float* b_out  = (const float*)d_in[4];   // [1]
    float* out = (float*)d_out;                    // [65536]

    char* ws = (char*)d_ws;
    const size_t wt_bytes   = (size_t)9 * NGATES * KPAD * 2;       //   442,368
    const size_t comb_bytes = (size_t)HW * KPAD * 2;               // 12,582,912
    _Float16* Wt    = (_Float16*)(ws);
    _Float16* comb0 = (_Float16*)(ws + ((wt_bytes + 511) & ~511ull));
    _Float16* comb1 = (_Float16*)((char*)comb0 + comb_bytes);
    float*    cstat = (float*)   ((char*)comb1 + comb_bytes);      // [HW][64] f32

    clstm_prep_w<<<(9 * NGATES * KPAD + 255) / 256, 256, 0, stream>>>(W_conv, Wt);
    clstm_init<<<HW / 256, 256, 0, stream>>>(cstat, comb0);

    _Float16* combs[2] = { comb0, comb1 };
    for (int t = 0; t < 16; ++t) {
        _Float16* cur = combs[t & 1];
        _Float16* nxt = combs[(t + 1) & 1];
        clstm_pack_x<<<HW / 256, 256, 0, stream>>>(seq + (size_t)t * 4 * HW, cur);
        clstm_step<<<HW / MTILE, 128, 0, stream>>>(cur, Wt, b_conv, cstat, nxt);
    }
    // after 16 steps, h lives in comb0 (16 % 2 == 0)
    clstm_out<<<HW / 256, 256, 0, stream>>>(comb0, W_out, b_out, out);
}
